// Local2DSum_23476291240799
// MI455X (gfx1250) — matrix operand order repaired and checked by reading, not compiled, not run on hardware
//
#include <hip/hip_runtime.h>

#define B_DIM 64
#define V_DIM 32
#define H_DIM 32
#define C_DIM 256
#define S_DIM 256
#define VH    (V_DIM * H_DIM)     // 1024
#define NHALF 128                 // S columns handled per workgroup
#define PAD   8                   // halves of padding per LDS row (keeps 16B align)
#define LROW  (C_DIM + PAD)       // 264 halves per LDS row

typedef __attribute__((ext_vector_type(16))) _Float16 v16h;
typedef __attribute__((ext_vector_type(8)))  _Float16 v8h;
typedef __attribute__((ext_vector_type(4)))  _Float16 v4h;
typedef __attribute__((ext_vector_type(8)))  float    v8f;
typedef __attribute__((ext_vector_type(4)))  float    v4f;

__global__ __launch_bounds__(256)
void logmm_wmma_kernel(const float* __restrict__ x,
                       const float* __restrict__ acc,
                       float* __restrict__ out)
{
    __shared__ __align__(16) _Float16 As[B_DIM][LROW];   // ex  : [b][c]  (f16)
    __shared__ __align__(16) _Float16 Bs[NHALF][LROW];   // ewT : [s][c]  (f16)
    __shared__ float mxs[B_DIM];
    __shared__ float mws[NHALF];
    __shared__ float redm[256];
    __shared__ float zpart[256];

    const int t     = threadIdx.x;
    const int gid   = blockIdx.x;
    const int vh    = gid >> 1;        // 0..1023
    const int nhalf = gid & 1;         // which half of S
    const int s0    = nhalf * NHALF;

    const float* accp = acc + (size_t)vh * (C_DIM * S_DIM);

    // ---------------- B tile: per-column max/Z over Cin, ew -> LDS (transposed) ----
    {
        const int sl    = t & (NHALF - 1);   // local s column 0..127
        const int chalf = t >> 7;            // 0 or 1 : which half of Cin
        const int c0    = chalf * (C_DIM / 2);
        const float* colp = accp + s0 + sl;

        float m = -__builtin_inff();
        #pragma unroll 4
        for (int c = 0; c < C_DIM / 2; ++c)
            m = fmaxf(m, colp[(size_t)(c0 + c) * S_DIM]);
        redm[t] = m;
        __syncthreads();

        const float cm = fmaxf(redm[sl], redm[sl + NHALF]);
        float z = 0.0f;
        #pragma unroll 4
        for (int c = 0; c < C_DIM / 2; ++c) {
            float e = __expf(colp[(size_t)(c0 + c) * S_DIM] - cm);
            z += e;
            Bs[sl][c0 + c] = (_Float16)e;    // transposed store: row = s, col = c
        }
        zpart[t] = z;
        __syncthreads();
        if (t < NHALF)
            mws[t] = -__logf(zpart[t] + zpart[t + NHALF]);
    }

    // ---------------- A tile: per-row max over Cin, ex -> LDS ----------------------
    {
        const int r = t >> 2;                // row b 0..63
        const int q = t & 3;                 // quarter-of-row lane group
        const float* xrow = x + ((size_t)r * VH + vh) * C_DIM;

        float m = -__builtin_inff();
        #pragma unroll
        for (int i = 0; i < 16; ++i) {
            v4f xv = *(const v4f*)(xrow + 4 * (q + 4 * i));
            m = fmaxf(fmaxf(fmaxf(m, xv.x), xv.y), fmaxf(xv.z, xv.w));
        }
        // 4 lanes share a row -> butterfly reduce
        m = fmaxf(m, __shfl_xor(m, 1, 32));
        m = fmaxf(m, __shfl_xor(m, 2, 32));

        #pragma unroll
        for (int i = 0; i < 16; ++i) {
            const int c = 4 * (q + 4 * i);
            v4f xv = *(const v4f*)(xrow + c);
            v4h ev;
            ev.x = (_Float16)__expf(xv.x - m);
            ev.y = (_Float16)__expf(xv.y - m);
            ev.z = (_Float16)__expf(xv.z - m);
            ev.w = (_Float16)__expf(xv.w - m);
            *(v4h*)&As[r][c] = ev;
        }
        if (q == 0) mxs[r] = m;
    }
    __syncthreads();

    // ---------------- WMMA GEMM from LDS ------------------------------------------
    const int lane  = t & 31;
    const int w     = t >> 5;      // wave 0..7
    const int mrow  = w & 3;       // M tile 0..3   (16 rows each)
    const int npair = w >> 2;      // N group 0..1  (64 cols each)
    const int lm    = lane & 15;
    const int lhi   = lane >> 4;   // half-wave select

    v8f accv[4];
    #pragma unroll
    for (int j = 0; j < 4; ++j) accv[j] = (v8f)0.0f;

    const int am = mrow * 16 + lm; // A row this lane feeds (same for both halves)

    #pragma unroll
    for (int kk = 0; kk < C_DIM / 32; ++kk) {
        // A fragment (16x32 f16): lane half selects K-offset 0/16; chunks at +0 / +8
        const int klo = kk * 32 + lhi * 16;
        v8h a0 = *(const v8h*)&As[am][klo];
        v8h a1 = *(const v8h*)&As[am][klo + 8];
        v16h af = __builtin_shufflevector(a0, a1,
                    0,1,2,3,4,5,6,7,8,9,10,11,12,13,14,15);

        const int kb = kk * 32 + lhi * 16;  // B: lanes 0-15 K0..15, lanes 16-31 K16..31
        #pragma unroll
        for (int j = 0; j < 4; ++j) {
            const int n = npair * 64 + j * 16 + lm;   // column (row of transposed LDS)
            v8h b0 = *(const v8h*)&Bs[n][kb];
            v8h b1 = *(const v8h*)&Bs[n][kb + 8];
            v16h bf = __builtin_shufflevector(b0, b1,
                        0,1,2,3,4,5,6,7,8,9,10,11,12,13,14,15);

            accv[j] = __builtin_amdgcn_wmma_f32_16x16x32_f16(
                          false, af, false, bf, (short)0, accv[j], false, false);
        }
    }

    // ---------------- epilogue: mx + mw + log(prod), write [B,V,H,S] ---------------
    #pragma unroll
    for (int j = 0; j < 4; ++j) {
        const int sl   = npair * 64 + j * 16 + lm;  // local column
        const int sg   = s0 + sl;                   // global s
        const float bw = mws[sl];
        #pragma unroll
        for (int r = 0; r < 8; ++r) {
            const int bl = mrow * 16 + lhi * 8 + r; // local/global b row
            const float val = mxs[bl] + bw + __logf(accv[j][r]);
            out[((size_t)bl * VH + vh) * S_DIM + sg] = val;
        }
    }
}

extern "C" void kernel_launch(void* const* d_in, const int* in_sizes, int n_in,
                              void* d_out, int out_size, void* d_ws, size_t ws_size,
                              hipStream_t stream) {
    const float* x   = (const float*)d_in[0];  // [B,V,H,Cin] f32
    const float* acc = (const float*)d_in[1];  // [V,H,Cin,S] f32
    float* out = (float*)d_out;                // [B,V,H,S]   f32

    dim3 grid(VH * 2);   // (v,h) x N-half
    dim3 block(256);     // 8 waves
    logmm_wmma_kernel<<<grid, block, 0, stream>>>(x, acc, out);
}
